// SheetMusicTeacherGAT_56023553409652
// MI455X (gfx1250) — matrix operand order repaired and checked by reading, not compiled, not run on hardware
//
#include <hip/hip_runtime.h>
#include <hip/hip_bf16.h>

// ---------------------------------------------------------------------------
// Problem constants (from the reference)
// ---------------------------------------------------------------------------
#define NN   30000
#define EE   480000
#define ET   (EE + NN)      // edges + self loops = 510000
#define GG   64
#define IN_F 134
#define IN_PAD 160          // 134 padded to multiple of 32 for WMMA K loop
#define H1C1 512            // 4 heads * 128
#define H2C2 256            // 1 head * 256

typedef __attribute__((ext_vector_type(16))) __bf16 bf16x16;
typedef __attribute__((ext_vector_type(8)))  float  f32x8;

// ---------------------------------------------------------------------------
// Helpers
// ---------------------------------------------------------------------------
__device__ __forceinline__ unsigned f2ord(float f) {
  unsigned u = __float_as_uint(f);
  return (u & 0x80000000u) ? ~u : (u | 0x80000000u);
}
__device__ __forceinline__ float ord2f(unsigned u) {
  return (u & 0x80000000u) ? __uint_as_float(u & 0x7fffffffu)
                           : __uint_as_float(~u);
}
__device__ __forceinline__ float wave_sum(float s) {
  for (int off = 16; off > 0; off >>= 1) s += __shfl_xor(s, off, 32);
  return s;
}

// ---------------------------------------------------------------------------
// Zero fill (for atomics targets; ordered-uint 0 == smaller than f2ord(-inf))
// ---------------------------------------------------------------------------
__global__ void k_fill0(unsigned* __restrict__ p, size_t n) {
  size_t i = (size_t)blockIdx.x * blockDim.x + threadIdx.x;
  if (i < n) p[i] = 0u;
}

// ---------------------------------------------------------------------------
// Feature assembly: [x_cont | class_emb | pitch_emb | 0 pad] -> bf16 [N,160]
// ---------------------------------------------------------------------------
__global__ void k_build_x(const float* __restrict__ xc,
                          const int* __restrict__ xcl,
                          const int* __restrict__ xpi,
                          const float* __restrict__ ctab,
                          const float* __restrict__ ptab,
                          __bf16* __restrict__ xb) {
  int i = blockIdx.x * blockDim.x + threadIdx.x;
  if (i >= NN * IN_PAD) return;
  int n = i / IN_PAD, c = i - n * IN_PAD;
  float v = 0.f;
  if (c < 6)        v = xc[n * 6 + c];
  else if (c < 70)  v = ctab[xcl[n] * 64 + (c - 6)];
  else if (c < 134) v = ptab[xpi[n] * 64 + (c - 70)];
  xb[i] = (__bf16)v;
}

// W[K,Nout] f32 -> BT[Nout,Kpad] bf16 (zero padded K)
__global__ void k_wt(const float* __restrict__ W, int K, int Nout,
                     __bf16* __restrict__ BT, int Kpad) {
  int i = blockIdx.x * blockDim.x + threadIdx.x;
  if (i >= Nout * Kpad) return;
  int n = i / Kpad, k = i - n * Kpad;
  BT[i] = (__bf16)(k < K ? W[(size_t)k * Nout + n] : 0.f);
}

__global__ void k_cvt_bf16(const float* __restrict__ in,
                           __bf16* __restrict__ out, int n) {
  int i = blockIdx.x * blockDim.x + threadIdx.x;
  if (i < n) out[i] = (__bf16)in[i];
}

// ---------------------------------------------------------------------------
// WMMA GEMM: C[M,Nout] = A[M,Kpad](bf16) * BT[Nout,Kpad]^T(bf16) + bias
// One wave computes a 16x64 tile. Per k-step: load A fragment once, load the
// four B fragments into DISTINCT register banks (so all loads issue in one
// clause), then drain four independent WMMAs back-to-back.
// ---------------------------------------------------------------------------
__global__ void k_gemm_wmma(const __bf16* __restrict__ A, int lda,
                            const __bf16* __restrict__ BT, int ldb,
                            const float* __restrict__ bias,
                            float* __restrict__ C, int ldc, int K) {
  int lane = threadIdx.x & 31;
  int half = lane >> 4;      // K-half selector
  int l16  = lane & 15;      // row of A / col of B
  int rowBase = blockIdx.y << 4;
  int colBase = blockIdx.x << 6;   // 64 output columns per wave
  const __bf16* arow  = A  + (size_t)(rowBase + l16) * lda;
  const __bf16* bbase = BT + (size_t)(colBase + l16) * ldb;
  f32x8 acc[4] = {{}, {}, {}, {}};
  for (int k = 0; k < K; k += 32) {
    bf16x16 a;
#pragma unroll
    for (int v = 0; v < 8; ++v) {
      // A 16x32 bf16 layout: VGPR v, low lanes K=(v/4)*16+(v%4)*2, high +8
      int ka = k + ((v >> 2) << 4) + ((v & 3) << 1) + (half << 3);
      a[2 * v]     = arow[ka];
      a[2 * v + 1] = arow[ka + 1];
    }
    bf16x16 bfr[4];
#pragma unroll
    for (int t = 0; t < 4; ++t) {
      const __bf16* brow = bbase + (size_t)(t << 4) * ldb;
#pragma unroll
      for (int v = 0; v < 8; ++v) {
        // B 32x16 bf16 layout: VGPR v, low lanes K=(v/4)*16+(v%4)*2, high +8
        int kb = k + ((v >> 2) << 4) + ((v & 3) << 1) + (half << 3);
        bfr[t][2 * v]     = brow[kb];
        bfr[t][2 * v + 1] = brow[kb + 1];
      }
    }
#pragma unroll
    for (int t = 0; t < 4; ++t) {
      acc[t] = __builtin_amdgcn_wmma_f32_16x16x32_bf16(
          false, a, false, bfr[t], (short)0, acc[t], false, false);
    }
  }
#pragma unroll
  for (int t = 0; t < 4; ++t) {
#pragma unroll
    for (int v = 0; v < 8; ++v) {
      int r  = rowBase + v + (half << 3);   // C: VGPR v -> M = v (+8 high half)
      int cI = colBase + (t << 4) + l16;
      C[(size_t)r * ldc + cI] = acc[t][v] + bias[cI];
    }
  }
}

// ---------------------------------------------------------------------------
// GATv2 attention passes
// ---------------------------------------------------------------------------
// pass 1: e = sum_c leakyrelu(xl[src]+xr[dst]) * att ; segment-max via atomicMax
__global__ void k_gat_scores(const float* __restrict__ xl,
                             const float* __restrict__ xr,
                             const float* __restrict__ att,
                             const int* __restrict__ ei,
                             int H, int C,
                             float* __restrict__ e_out,
                             unsigned* __restrict__ uemax) {
  int w    = (blockIdx.x * blockDim.x + threadIdx.x) >> 5;
  int lane = threadIdx.x & 31;
  int total = ET * H;
  if (w >= total) return;
  int edge = w / H, head = w - edge * H;
  int s = edge < EE ? ei[edge]      : edge - EE;
  int d = edge < EE ? ei[EE + edge] : edge - EE;
  const float* pl = xl + ((size_t)s * H + head) * C;
  const float* pr = xr + ((size_t)d * H + head) * C;
  const float* pa = att + head * C;
  float sum = 0.f;
  for (int i = lane; i < C; i += 32) {
    float v = pl[i] + pr[i];
    v = v > 0.f ? v : 0.2f * v;   // leaky_relu slope 0.2
    sum += v * pa[i];
  }
  sum = wave_sum(sum);
  if (lane == 0) {
    e_out[w] = sum;
    atomicMax(&uemax[(size_t)d * H + head], f2ord(sum));
  }
}

// pass 2: ex = exp(e - emax[dst]); den[dst] += ex
__global__ void k_gat_expsum(float* __restrict__ e_io,
                             const unsigned* __restrict__ uemax,
                             float* __restrict__ den,
                             const int* __restrict__ ei, int H) {
  int w = blockIdx.x * blockDim.x + threadIdx.x;
  if (w >= ET * H) return;
  int edge = w / H, head = w - edge * H;
  int d = edge < EE ? ei[EE + edge] : edge - EE;
  float ex = expf(e_io[w] - ord2f(uemax[(size_t)d * H + head]));
  e_io[w] = ex;
  atomicAdd(&den[(size_t)d * H + head], ex);
}

// pass 3: out[dst] += (ex/den[dst]) * xl[src]
__global__ void k_gat_scatter(const float* __restrict__ e_ex,
                              const float* __restrict__ den,
                              const float* __restrict__ xl,
                              const int* __restrict__ ei,
                              int H, int C,
                              float* __restrict__ out) {
  int w    = (blockIdx.x * blockDim.x + threadIdx.x) >> 5;
  int lane = threadIdx.x & 31;
  if (w >= ET * H) return;
  int edge = w / H, head = w - edge * H;
  int s = edge < EE ? ei[edge]      : edge - EE;
  int d = edge < EE ? ei[EE + edge] : edge - EE;
  float alpha = e_ex[w] / (den[(size_t)d * H + head] + 1e-16f);
  const float* pl = xl + ((size_t)s * H + head) * C;
  float* po       = out + ((size_t)d * H + head) * C;
  for (int i = lane; i < C; i += 32) atomicAdd(&po[i], alpha * pl[i]);
}

// ---------------------------------------------------------------------------
// LayerNorm((in + bias)) * g + b  then activation. One wave per row.
// act: 0 = ELU(alpha=1), 1 = exact GELU
// ---------------------------------------------------------------------------
__global__ void k_ln_act(const float* __restrict__ in,
                         const float* __restrict__ bias,
                         const float* __restrict__ g,
                         const float* __restrict__ b,
                         float* __restrict__ out, int D, int rows, int act) {
  int w    = (blockIdx.x * blockDim.x + threadIdx.x) >> 5;
  int lane = threadIdx.x & 31;
  if (w >= rows) return;
  const float* p = in + (size_t)w * D;
  float s = 0.f;
  for (int i = lane; i < D; i += 32) s += p[i] + bias[i];
  s = wave_sum(s);
  float mean = s / (float)D;
  float vs = 0.f;
  for (int i = lane; i < D; i += 32) {
    float t = p[i] + bias[i] - mean;
    vs += t * t;
  }
  vs = wave_sum(vs);
  float inv = rsqrtf(vs / (float)D + 1e-5f);
  float* o = out + (size_t)w * D;
  for (int i = lane; i < D; i += 32) {
    float y = (p[i] + bias[i] - mean) * inv * g[i] + b[i];
    if (act == 0) y = y > 0.f ? y : (expf(y) - 1.f);                  // ELU
    else          y = 0.5f * y * (1.f + erff(y * 0.70710678118f));    // GELU
    o[i] = y;
  }
}

// ---------------------------------------------------------------------------
// Global mean pool
// ---------------------------------------------------------------------------
__global__ void k_pool_sum(const float* __restrict__ x,
                           const int* __restrict__ batch,
                           float* __restrict__ sums,
                           float* __restrict__ cnt) {
  int i = blockIdx.x * blockDim.x + threadIdx.x;
  if (i >= NN * H2C2) return;
  int n = i / H2C2, c = i - n * H2C2;
  int gI = batch[n];
  atomicAdd(&sums[(size_t)gI * H2C2 + c], x[i]);
  if (c == 0) atomicAdd(&cnt[gI], 1.f);
}

__global__ void k_pool_div(const float* __restrict__ sums,
                           const float* __restrict__ cnt,
                           float* __restrict__ pooled) {
  int i = blockIdx.x * blockDim.x + threadIdx.x;
  if (i >= GG * H2C2) return;
  int gI = i / H2C2;
  pooled[i] = sums[i] / fmaxf(cnt[gI], 1.f);
}

// ---------------------------------------------------------------------------
// Small dense layer (64 rows): out = A @ W + bias
// ---------------------------------------------------------------------------
__global__ void k_dense(const float* __restrict__ A,
                        const float* __restrict__ W,
                        const float* __restrict__ bias,
                        float* __restrict__ out, int M, int K, int Nout) {
  int i = blockIdx.x * blockDim.x + threadIdx.x;
  if (i >= M * Nout) return;
  int m = i / Nout, j = i - m * Nout;
  float s = bias[j];
  const float* a = A + (size_t)m * K;
  for (int k = 0; k < K; ++k) s += a[k] * W[(size_t)k * Nout + j];
  out[i] = s;
}

// ---------------------------------------------------------------------------
// Row-wise L2 normalize, one wave per row
// ---------------------------------------------------------------------------
__global__ void k_l2norm(const float* __restrict__ in,
                         float* __restrict__ out, int D, int rows) {
  int w    = (blockIdx.x * blockDim.x + threadIdx.x) >> 5;
  int lane = threadIdx.x & 31;
  if (w >= rows) return;
  const float* p = in + (size_t)w * D;
  float ss = 0.f;
  for (int i = lane; i < D; i += 32) ss += p[i] * p[i];
  ss = wave_sum(ss);
  float inv = 1.f / fmaxf(sqrtf(ss), 1e-12f);
  float* o = out + (size_t)w * D;
  for (int i = lane; i < D; i += 32) o[i] = p[i] * inv;
}

// ---------------------------------------------------------------------------
// Host side
// ---------------------------------------------------------------------------
static inline int cdiv(long long a, long long b) { return (int)((a + b - 1) / b); }

extern "C" void kernel_launch(void* const* d_in, const int* in_sizes, int n_in,
                              void* d_out, int out_size, void* d_ws, size_t ws_size,
                              hipStream_t stream) {
  const float* x_cont   = (const float*)d_in[0];
  const int*   x_class  = (const int*)  d_in[1];
  const int*   x_pitch  = (const int*)  d_in[2];
  const int*   edge_idx = (const int*)  d_in[3];
  const int*   batch    = (const int*)  d_in[4];
  const float* class_tab= (const float*)d_in[5];
  const float* pitch_tab= (const float*)d_in[6];
  const float* Wl1 = (const float*)d_in[7];
  const float* bl1 = (const float*)d_in[8];
  const float* Wr1 = (const float*)d_in[9];
  const float* br1 = (const float*)d_in[10];
  const float* att1 = (const float*)d_in[11];
  const float* bias1= (const float*)d_in[12];
  const float* ln1g = (const float*)d_in[13];
  const float* ln1b = (const float*)d_in[14];
  const float* Wl2 = (const float*)d_in[15];
  const float* bl2 = (const float*)d_in[16];
  const float* Wr2 = (const float*)d_in[17];
  const float* br2 = (const float*)d_in[18];
  const float* att2 = (const float*)d_in[19];
  const float* bias2= (const float*)d_in[20];
  const float* ln2g = (const float*)d_in[21];
  const float* ln2b = (const float*)d_in[22];
  const float* Wp1 = (const float*)d_in[23];
  const float* bp1 = (const float*)d_in[24];
  const float* lnpg= (const float*)d_in[25];
  const float* lnpb= (const float*)d_in[26];
  const float* Wp2 = (const float*)d_in[27];
  const float* bp2 = (const float*)d_in[28];

  // ---- workspace layout ----
  char* base = (char*)d_ws;
  size_t off = 0;
  auto alloc = [&](size_t bytes) -> void* {
    void* p = base + off;
    off = (off + bytes + 255) & ~(size_t)255;
    return p;
  };
  __bf16* xb    = (__bf16*)alloc((size_t)NN * IN_PAD * 2);
  __bf16* WlT1  = (__bf16*)alloc((size_t)H1C1 * IN_PAD * 2);
  __bf16* WrT1  = (__bf16*)alloc((size_t)H1C1 * IN_PAD * 2);
  float*  xl1   = (float*)alloc((size_t)NN * H1C1 * 4);
  float*  xr1   = (float*)alloc((size_t)NN * H1C1 * 4);
  float*  e1    = (float*)alloc((size_t)ET * 4 * 4);
  unsigned* um1 = (unsigned*)alloc((size_t)NN * 4 * 4);
  float*  den1  = (float*)alloc((size_t)NN * 4 * 4);
  float*  out1  = (float*)alloc((size_t)NN * H1C1 * 4);
  float*  x2    = (float*)alloc((size_t)NN * H1C1 * 4);
  __bf16* x2b   = (__bf16*)alloc((size_t)NN * H1C1 * 2);
  __bf16* WlT2  = (__bf16*)alloc((size_t)H2C2 * H1C1 * 2);
  __bf16* WrT2  = (__bf16*)alloc((size_t)H2C2 * H1C1 * 2);
  float*  xl2   = (float*)alloc((size_t)NN * H2C2 * 4);
  float*  xr2   = (float*)alloc((size_t)NN * H2C2 * 4);
  float*  e2    = (float*)alloc((size_t)ET * 4);
  unsigned* um2 = (unsigned*)alloc((size_t)NN * 4);
  float*  den2  = (float*)alloc((size_t)NN * 4);
  float*  out2  = (float*)alloc((size_t)NN * H2C2 * 4);
  float*  x3    = (float*)alloc((size_t)NN * H2C2 * 4);
  float*  sums  = (float*)alloc((size_t)GG * H2C2 * 4);
  float*  cnt   = (float*)alloc((size_t)GG * 4);
  float*  pooled= (float*)alloc((size_t)GG * H2C2 * 4);
  float*  tmp1  = (float*)alloc((size_t)GG * 512 * 4);
  float*  hbuf  = (float*)alloc((size_t)GG * 512 * 4);
  float*  oraw  = (float*)alloc((size_t)GG * 512 * 4);
  float*  zb512 = (float*)alloc(512 * 4);

  const int TB = 256;
  auto Z = [&](void* p, size_t nwords) {
    k_fill0<<<cdiv((long long)nwords, TB), TB, 0, stream>>>((unsigned*)p, nwords);
  };
  // zero-init all atomic targets + zero-bias vector
  Z(um1, (size_t)NN * 4);  Z(den1, (size_t)NN * 4);  Z(out1, (size_t)NN * H1C1);
  Z(um2, NN);              Z(den2, NN);              Z(out2, (size_t)NN * H2C2);
  Z(sums, (size_t)GG * H2C2); Z(cnt, GG);            Z(zb512, 512);

  // ---- features + weight prep ----
  k_build_x<<<cdiv((long long)NN * IN_PAD, TB), TB, 0, stream>>>(
      x_cont, x_class, x_pitch, class_tab, pitch_tab, xb);
  k_wt<<<cdiv((long long)H1C1 * IN_PAD, TB), TB, 0, stream>>>(Wl1, IN_F, H1C1, WlT1, IN_PAD);
  k_wt<<<cdiv((long long)H1C1 * IN_PAD, TB), TB, 0, stream>>>(Wr1, IN_F, H1C1, WrT1, IN_PAD);
  k_wt<<<cdiv((long long)H2C2 * H1C1, TB), TB, 0, stream>>>(Wl2, H1C1, H2C2, WlT2, H1C1);
  k_wt<<<cdiv((long long)H2C2 * H1C1, TB), TB, 0, stream>>>(Wr2, H1C1, H2C2, WrT2, H1C1);

  // ---- layer 1: WMMA GEMMs (16x64 tile per wave) ----
  {
    dim3 grid(H1C1 / 64, NN / 16);
    k_gemm_wmma<<<grid, 32, 0, stream>>>(xb, IN_PAD, WlT1, IN_PAD, bl1, xl1, H1C1, IN_PAD);
    k_gemm_wmma<<<grid, 32, 0, stream>>>(xb, IN_PAD, WrT1, IN_PAD, br1, xr1, H1C1, IN_PAD);
  }
  // ---- layer 1: attention ----
  {
    long long waves = (long long)ET * 4;
    k_gat_scores<<<cdiv(waves * 32, TB), TB, 0, stream>>>(xl1, xr1, att1, edge_idx, 4, 128, e1, um1);
    k_gat_expsum<<<cdiv(waves, TB), TB, 0, stream>>>(e1, um1, den1, edge_idx, 4);
    k_gat_scatter<<<cdiv(waves * 32, TB), TB, 0, stream>>>(e1, den1, xl1, edge_idx, 4, 128, out1);
  }
  // LN + ELU -> x2
  k_ln_act<<<cdiv((long long)NN * 32, TB), TB, 0, stream>>>(out1, bias1, ln1g, ln1b, x2, H1C1, NN, 0);

  // ---- layer 2: WMMA GEMMs ----
  k_cvt_bf16<<<cdiv((long long)NN * H1C1, TB), TB, 0, stream>>>(x2, x2b, NN * H1C1);
  {
    dim3 grid(H2C2 / 64, NN / 16);
    k_gemm_wmma<<<grid, 32, 0, stream>>>(x2b, H1C1, WlT2, H1C1, bl2, xl2, H2C2, H1C1);
    k_gemm_wmma<<<grid, 32, 0, stream>>>(x2b, H1C1, WrT2, H1C1, br2, xr2, H2C2, H1C1);
  }
  // ---- layer 2: attention (H=1, C=256) ----
  {
    long long waves = (long long)ET;
    k_gat_scores<<<cdiv(waves * 32, TB), TB, 0, stream>>>(xl2, xr2, att2, edge_idx, 1, 256, e2, um2);
    k_gat_expsum<<<cdiv(waves, TB), TB, 0, stream>>>(e2, um2, den2, edge_idx, 1);
    k_gat_scatter<<<cdiv(waves * 32, TB), TB, 0, stream>>>(e2, den2, xl2, edge_idx, 1, 256, out2);
  }
  // LN + ELU -> x3
  k_ln_act<<<cdiv((long long)NN * 32, TB), TB, 0, stream>>>(out2, bias2, ln2g, ln2b, x3, H2C2, NN, 0);

  // ---- global mean pool ----
  k_pool_sum<<<cdiv((long long)NN * H2C2, TB), TB, 0, stream>>>(x3, batch, sums, cnt);
  k_pool_div<<<cdiv((long long)GG * H2C2, TB), TB, 0, stream>>>(sums, cnt, pooled);

  // ---- projection head ----
  k_dense<<<cdiv((long long)GG * 512, TB), TB, 0, stream>>>(pooled, Wp1, bp1, tmp1, GG, H2C2, 512);
  k_ln_act<<<cdiv((long long)GG * 32, TB), TB, 0, stream>>>(tmp1, zb512, lnpg, lnpb, hbuf, 512, GG, 1);
  k_dense<<<cdiv((long long)GG * 512, TB), TB, 0, stream>>>(hbuf, Wp2, bp2, oraw, GG, 512, 512);
  k_l2norm<<<cdiv((long long)GG * 32, TB), TB, 0, stream>>>(oraw, (float*)d_out, 512, GG);

  (void)n_in; (void)in_sizes; (void)out_size; (void)ws_size;
}
